// MultiHeadAttention_38122129719368
// MI455X (gfx1250) — compile-verified, matrix operand order checked by reading, not run on hardware
//
#include <hip/hip_runtime.h>

typedef __attribute__((ext_vector_type(8)))  unsigned short v8u;
typedef __attribute__((ext_vector_type(16))) unsigned short v16u;
typedef __attribute__((ext_vector_type(16))) __bf16        v16bf;
typedef __attribute__((ext_vector_type(8)))  float         v8f;

constexpr int BATCH = 4;
constexpr int SEQ   = 2048;
constexpr int DM    = 1024;
constexpr int NH    = 16;
constexpr int HD    = 64;
constexpr int MTOT  = BATCH * SEQ;   // 8192

// padded LDS row strides (odd 16B-chunk counts -> conflict-free-ish, 16B aligned)
constexpr int KSTR = 72;   // K tile row stride (elements), 144B
constexpr int VSTR = 40;   // V tile row stride (elements), 80B

__device__ __forceinline__ unsigned short f2bf(float f) {
  unsigned u = __float_as_uint(f);
  u += 0x7FFFu + ((u >> 16) & 1u);   // round-to-nearest-even
  return (unsigned short)(u >> 16);
}

// Load 16 bf16 elements in the CDNA5 16-bit WMMA per-lane K pattern:
// 8 contiguous at p, 8 contiguous at p+16 (both 16B-aligned b128 loads).
__device__ __forceinline__ v16bf load16(const unsigned short* p) {
  union { v16u v; v8u h[2]; } u;
  u.h[0] = *(const v8u*)(p);
  u.h[1] = *(const v8u*)(p + 16);
  return __builtin_bit_cast(v16bf, u.v);
}

__device__ __forceinline__ v8f wmma_bf16(v16bf a, v16bf b, v8f c) {
  return __builtin_amdgcn_wmma_f32_16x16x32_bf16(false, a, false, b, (short)0, c,
                                                 false, false);
}

// async 16B copy global -> LDS (gfx1250), tracked by ASYNCcnt
__device__ __forceinline__ void async_cp16(unsigned ldsOff, const unsigned short* g) {
  asm volatile("global_load_async_to_lds_b128 %0, %1, off"
               :: "v"(ldsOff), "v"(g) : "memory");
}
__device__ __forceinline__ void wait_async0() {
  asm volatile("s_wait_asynccnt 0" ::: "memory");
}

__global__ void cvt_f32_bf16(const float* __restrict__ in,
                             unsigned short* __restrict__ out, int n) {
  int i = blockIdx.x * blockDim.x + threadIdx.x;
  if (i < n) out[i] = f2bf(in[i]);
}

// y[m,n] = sum_k A[m,k] * W[n,k]   (A: [M,1024] bf16 rm, W: [1024,1024] bf16 rm)
// mode 0: write bf16 to [b,h,t,d]     (Q and K)
// mode 2: write bf16 to [b,h,d,t]     (V transposed)
// mode 3: write fp32 y + bias to [m,n] (output projection)
__global__ __launch_bounds__(256) void gemm_bf16(
    const unsigned short* __restrict__ A,
    const unsigned short* __restrict__ W,
    unsigned short* __restrict__ outb,
    float* __restrict__ outf,
    const float* __restrict__ bias,
    int mode) {
  const int lane = threadIdx.x & 31;
  const int wave = threadIdx.x >> 5;
  const int wm = wave >> 2, wn = wave & 3;          // 2x4 wave grid
  const int tileM = blockIdx.x * 64 + wm * 32;
  const int tileN = blockIdx.y * 128 + wn * 32;
  const int hf = lane >> 4, l16 = lane & 15, kOff = hf * 8;

  const unsigned short* a0p = A + (size_t)(tileM + l16) * DM + kOff;
  const unsigned short* a1p = A + (size_t)(tileM + 16 + l16) * DM + kOff;
  const unsigned short* b0p = W + (size_t)(tileN + l16) * DM + kOff;
  const unsigned short* b1p = W + (size_t)(tileN + 16 + l16) * DM + kOff;

  v8f acc[2][2] = {};
  for (int k = 0; k < DM; k += 32) {
    __builtin_prefetch(a0p + k + 128, 0, 3);
    __builtin_prefetch(b0p + k + 128, 0, 3);
    v16bf a0 = load16(a0p + k);
    v16bf a1 = load16(a1p + k);
    v16bf b0 = load16(b0p + k);
    v16bf b1 = load16(b1p + k);
    acc[0][0] = wmma_bf16(a0, b0, acc[0][0]);
    acc[0][1] = wmma_bf16(a0, b1, acc[0][1]);
    acc[1][0] = wmma_bf16(a1, b0, acc[1][0]);
    acc[1][1] = wmma_bf16(a1, b1, acc[1][1]);
  }

#pragma unroll
  for (int i = 0; i < 2; ++i)
#pragma unroll
    for (int j = 0; j < 2; ++j)
#pragma unroll
      for (int e = 0; e < 8; ++e) {
        float v = acc[i][j][e];
        int m = tileM + i * 16 + hf * 8 + e;
        int n = tileN + j * 16 + l16;
        if (mode == 3) {
          outf[(size_t)m * DM + n] = v + bias[n];
        } else {
          int b = m >> 11, t = m & (SEQ - 1);
          int h = n >> 6,  d = n & (HD - 1);
          size_t addr = (mode == 2)
              ? ((size_t)(b * NH + h) * HD + d) * SEQ + t
              : ((size_t)(b * NH + h) * SEQ + t) * HD + d;
          outb[addr] = f2bf(v);
        }
      }
}

// Flash attention, causal. Grid: (B*H, SEQ/64). Block: 128 (4 waves).
// K/V tiles async-DMA'd into double-buffered LDS shared by all 4 waves.
__global__ __launch_bounds__(128) void attn_fwd(
    const unsigned short* __restrict__ Qm,   // [bh][t][d]
    const unsigned short* __restrict__ Km,   // [bh][t][d]
    const unsigned short* __restrict__ Vt,   // [bh][d][t]
    unsigned short* __restrict__ Ctx) {      // [b][t][h*64+d]
  __shared__ __align__(16) unsigned short kbuf[2][32 * KSTR];
  __shared__ __align__(16) unsigned short vbuf[2][64 * VSTR];
  __shared__ __align__(16) unsigned short pbuf[4][16][40];

  const int bh = blockIdx.x;
  const int qt = blockIdx.y;
  const int tid  = threadIdx.x;
  const int wave = tid >> 5;
  const int lane = tid & 31;
  const int hf = lane >> 4, l16 = lane & 15, kOff = hf * 8;

  const unsigned short* Qb = Qm + (size_t)bh * SEQ * HD;
  const unsigned short* Kb = Km + (size_t)bh * SEQ * HD;
  const unsigned short* Vb = Vt + (size_t)bh * HD * SEQ;

  // cooperative async copy of one 32-key block of K (32x64) and V (64x32)
  auto issue_copy = [&](int kb, int stage) {
#pragma unroll
    for (int it = 0; it < 2; ++it) {       // K: 256 chunks of 16B
      int c = tid + it * 128;
      int row = c >> 3, col = c & 7;
      async_cp16((unsigned)(size_t)&kbuf[stage][row * KSTR + col * 8],
                 Kb + (size_t)(kb + row) * HD + col * 8);
    }
#pragma unroll
    for (int it = 0; it < 2; ++it) {       // V: 256 chunks of 16B
      int c = tid + it * 128;
      int row = c >> 2, col = c & 3;
      async_cp16((unsigned)(size_t)&vbuf[stage][row * VSTR + col * 8],
                 Vb + (size_t)row * SEQ + kb + col * 8);
    }
  };

  const int qRow0 = qt * 64 + wave * 16;
  const unsigned short* qrow = Qb + (size_t)(qRow0 + l16) * HD + kOff;
  v16bf aq0 = load16(qrow);        // d = 0..31 slice
  v16bf aq1 = load16(qrow + 32);   // d = 32..63 slice

  float rmax[8], rsum[8];
#pragma unroll
  for (int e = 0; e < 8; ++e) { rmax[e] = -1e30f; rsum[e] = 0.f; }
  v8f ctx[4] = {};

  const int kEnd = qt * 64 + 64;
  issue_copy(0, 0);
  wait_async0();
  __syncthreads();

  int stage = 0;
  for (int kb = 0; kb < kEnd; kb += 32) {
    if (kb + 32 < kEnd) issue_copy(kb + 32, stage ^ 1);   // overlap next DMA

    // S = Q K^T for two 16-key tiles, B operand from LDS
    v8f S[2];
#pragma unroll
    for (int j = 0; j < 2; ++j) {
      const unsigned short* krow = &kbuf[stage][(j * 16 + l16) * KSTR + kOff];
      v16bf bk0 = load16(krow);
      v16bf bk1 = load16(krow + 32);
      v8f s = {};
      s = wmma_bf16(aq0, bk0, s);
      s = wmma_bf16(aq1, bk1, s);
      S[j] = s;
    }
    // scale + causal mask
#pragma unroll
    for (int j = 0; j < 2; ++j)
#pragma unroll
      for (int e = 0; e < 8; ++e) {
        int q  = qRow0 + hf * 8 + e;
        int kc = kb + j * 16 + l16;
        float v = S[j][e] * 0.125f;   // 1/sqrt(64)
        S[j][e] = (kc > q) ? -3.0e38f : v;
      }
    // online softmax; rows live in 16-lane halves -> xor-butterfly over 8..1
#pragma unroll
    for (int e = 0; e < 8; ++e) {
      float m = fmaxf(S[0][e], S[1][e]);
#pragma unroll
      for (int off = 8; off >= 1; off >>= 1)
        m = fmaxf(m, __shfl_xor(m, off, 32));
      float nm = fmaxf(rmax[e], m);
      float scale = __expf(rmax[e] - nm);
      rmax[e] = nm;
      float p0 = __expf(S[0][e] - nm);
      float p1 = __expf(S[1][e] - nm);
      float rs = p0 + p1;
#pragma unroll
      for (int off = 8; off >= 1; off >>= 1)
        rs += __shfl_xor(rs, off, 32);
      rsum[e] = rsum[e] * scale + rs;
#pragma unroll
      for (int c = 0; c < 4; ++c) ctx[c][e] *= scale;
      int r = hf * 8 + e;                    // P bounce: C-layout -> A-layout
      pbuf[wave][r][l16]      = f2bf(p0);
      pbuf[wave][r][16 + l16] = f2bf(p1);
    }
    // re-load P in A-matrix layout (DS ops are in-order within a wave)
    union { v16u v; v8u h2[2]; } up;
    up.h2[0] = *(const v8u*)&pbuf[wave][l16][kOff];
    up.h2[1] = *(const v8u*)&pbuf[wave][l16][kOff + 16];
    v16bf ap = __builtin_bit_cast(v16bf, up.v);
    // ctx += P @ V, V operand from LDS (pre-transposed: contiguous along keys)
#pragma unroll
    for (int c = 0; c < 4; ++c) {
      const unsigned short* vrow = &vbuf[stage][(c * 16 + l16) * VSTR + kOff];
      v16bf bv = load16(vrow);
      ctx[c] = wmma_bf16(ap, bv, ctx[c]);
    }

    wait_async0();       // next stage's DMA complete (per-wave ASYNCcnt)
    __syncthreads();     // visible to all waves; prev buffer free to overwrite
    stage ^= 1;
  }

  const int b = bh >> 4, h = bh & (NH - 1);
#pragma unroll
  for (int c = 0; c < 4; ++c)
#pragma unroll
    for (int e = 0; e < 8; ++e) {
      int t = qRow0 + hf * 8 + e;
      int d = c * 16 + l16;
      float v = ctx[c][e] / rsum[e];
      Ctx[((size_t)(b * SEQ + t)) * DM + h * HD + d] = f2bf(v);
    }
}

extern "C" void kernel_launch(void* const* d_in, const int* in_sizes, int n_in,
                              void* d_out, int out_size, void* d_ws, size_t ws_size,
                              hipStream_t stream) {
  (void)in_sizes; (void)n_in; (void)out_size; (void)ws_size;
  const float* x  = (const float*)d_in[0];
  const float* Wq = (const float*)d_in[1];
  const float* Wk = (const float*)d_in[2];
  const float* Wv = (const float*)d_in[3];
  const float* Wo = (const float*)d_in[4];
  const float* bo = (const float*)d_in[5];

  char* ws = (char*)d_ws;
  size_t off = 0;
  auto alloc = [&](size_t bytes) { char* p = ws + off; off += (bytes + 255) & ~(size_t)255; return p; };
  unsigned short* xbf  = (unsigned short*)alloc((size_t)MTOT * DM * 2);
  unsigned short* wqb  = (unsigned short*)alloc((size_t)DM * DM * 2);
  unsigned short* wkb  = (unsigned short*)alloc((size_t)DM * DM * 2);
  unsigned short* wvb  = (unsigned short*)alloc((size_t)DM * DM * 2);
  unsigned short* wob  = (unsigned short*)alloc((size_t)DM * DM * 2);
  unsigned short* Qmw  = (unsigned short*)alloc((size_t)MTOT * DM * 2);
  unsigned short* Kmw  = (unsigned short*)alloc((size_t)MTOT * DM * 2);
  unsigned short* Vtw  = (unsigned short*)alloc((size_t)MTOT * DM * 2);
  unsigned short* ctxb = (unsigned short*)alloc((size_t)MTOT * DM * 2);

  const int nX = MTOT * DM;   // 8388608
  const int nW = DM * DM;     // 1048576
  cvt_f32_bf16<<<(nX + 255) / 256, 256, 0, stream>>>(x,  xbf, nX);
  cvt_f32_bf16<<<(nW + 255) / 256, 256, 0, stream>>>(Wq, wqb, nW);
  cvt_f32_bf16<<<(nW + 255) / 256, 256, 0, stream>>>(Wk, wkb, nW);
  cvt_f32_bf16<<<(nW + 255) / 256, 256, 0, stream>>>(Wv, wvb, nW);
  cvt_f32_bf16<<<(nW + 255) / 256, 256, 0, stream>>>(Wo, wob, nW);

  dim3 gg(MTOT / 64, DM / 128);
  gemm_bf16<<<gg, 256, 0, stream>>>(xbf, wqb, Qmw, nullptr, nullptr, 0);
  gemm_bf16<<<gg, 256, 0, stream>>>(xbf, wkb, Kmw, nullptr, nullptr, 0);
  gemm_bf16<<<gg, 256, 0, stream>>>(xbf, wvb, Vtw, nullptr, nullptr, 2);

  attn_fwd<<<dim3(BATCH * NH, SEQ / 64), 128, 0, stream>>>(Qmw, Kmw, Vtw, ctxb);

  gemm_bf16<<<gg, 256, 0, stream>>>(ctxb, wob, nullptr, (float*)d_out, bo, 3);
}
